// QuantumCNN_62697932587040
// MI455X (gfx1250) — matrix operand order repaired and checked by reading, not compiled, not run on hardware
//
#include <hip/hip_runtime.h>
#include <math.h>

typedef __attribute__((ext_vector_type(16))) _Float16 v16h;
typedef __attribute__((ext_vector_type(8)))  float    v8f;

#define NQ  6
#define DIM 64

#define H1_STRIDE 34   // 16x16 f16 rows padded: 34 halves (68B) -> no 64-bank wrap
#define H2_STRIDE 17   // f32 rows padded: 17 floats (68B)

// ---------------- quantum gates: state fully in registers ----------------
// Qubit q is bit (5-q) of the flat 64-index (reference layout).

template<int Q>
static __device__ __forceinline__ void gate_ry(float* sr, float* si, float c, float s) {
    constexpr int STR = 1 << (5 - Q);
#pragma unroll
    for (int b = 0; b < DIM; b += 2 * STR)
#pragma unroll
        for (int o = 0; o < STR; ++o) {
            const int i0 = b + o, i1 = i0 + STR;
            const float r0 = sr[i0], m0 = si[i0], r1 = sr[i1], m1 = si[i1];
            sr[i0] = c * r0 - s * r1;  si[i0] = c * m0 - s * m1;
            sr[i1] = s * r0 + c * r1;  si[i1] = s * m0 + c * m1;
        }
}

template<int Q>
static __device__ __forceinline__ void gate_rx(float* sr, float* si, float c, float s) {
    constexpr int STR = 1 << (5 - Q);
#pragma unroll
    for (int b = 0; b < DIM; b += 2 * STR)
#pragma unroll
        for (int o = 0; o < STR; ++o) {
            const int i0 = b + o, i1 = i0 + STR;
            const float r0 = sr[i0], m0 = si[i0], r1 = sr[i1], m1 = si[i1];
            // new0 = c*s0 - i*s*s1 ; new1 = -i*s*s0 + c*s1
            sr[i0] =  c * r0 + s * m1;  si[i0] = c * m0 - s * r1;
            sr[i1] =  s * m0 + c * r1;  si[i1] = -s * r0 + c * m1;
        }
}

template<int Q>
static __device__ __forceinline__ void gate_rz(float* sr, float* si, float c, float s) {
    constexpr int STR = 1 << (5 - Q);
#pragma unroll
    for (int b = 0; b < DIM; b += 2 * STR)
#pragma unroll
        for (int o = 0; o < STR; ++o) {
            const int i0 = b + o, i1 = i0 + STR;
            const float r0 = sr[i0], m0 = si[i0], r1 = sr[i1], m1 = si[i1];
            // s0 *= exp(-ih) = c - i s ; s1 *= exp(+ih) = c + i s
            sr[i0] = r0 * c + m0 * s;  si[i0] = m0 * c - r0 * s;
            sr[i1] = r1 * c - m1 * s;  si[i1] = m1 * c + r1 * s;
        }
}

template<int C, int T>
static __device__ __forceinline__ void gate_cnot(float* sr, float* si) {
    constexpr int CB = 1 << (5 - C);
    constexpr int TB = 1 << (5 - T);
#pragma unroll
    for (int k = 0; k < DIM; ++k) {
        if ((k & CB) && !(k & TB)) {
            const int j = k | TB;
            float t;
            t = sr[k]; sr[k] = sr[j]; sr[j] = t;
            t = si[k]; si[k] = si[j]; si[j] = t;
        }
    }
}

// ---------------- fused kernel ----------------
__global__ __launch_bounds__(256)
void qcnn_fused(const float* __restrict__ x,
                const float* __restrict__ W1, const float* __restrict__ b1,
                const float* __restrict__ W2, const float* __restrict__ b2,
                const float* __restrict__ W3, const float* __restrict__ b3,
                const float* __restrict__ qw,
                const float* __restrict__ W4, const float* __restrict__ b4,
                const float* __restrict__ W5, const float* __restrict__ b5,
                float* __restrict__ out, int B)
{
    __shared__ _Float16 W1h[128 * 32];
    __shared__ _Float16 W2h[32 * 16];
    __shared__ _Float16 h1h[256 * H1_STRIDE];   // relu(x@W1+b1), f16, padded rows
    __shared__ float    h2f[256 * H2_STRIDE];   // relu(h1@W2+b2), f32, padded rows

    const int tid  = threadIdx.x;
    const int lane = tid & 31;
    const int wave = tid >> 5;
    const int g    = lane >> 4;   // lane half
    const int ln   = lane & 15;
    const int blockBase = blockIdx.x * 256;

    // stage weights as f16
    for (int i = tid; i < 128 * 32; i += 256) W1h[i] = (_Float16)W1[i];
    for (int i = tid; i < 32 * 16;  i += 256) W2h[i] = (_Float16)W2[i];
    __syncthreads();

    // ---- Layer 1 (WMMA, K=128) : each wave computes 2 M-tiles of 16 rows ----
#pragma unroll
    for (int tt = 0; tt < 2; ++tt) {
        const int tile = wave * 2 + tt;            // 0..15 -> 256 rows/block
        int rowG = blockBase + tile * 16 + ln;
        if (rowG >= B) rowG = B - 1;               // clamp (load safety)
        const float* xr = x + (size_t)rowG * 128;

        // Build A-fragments for all 4 K-steps ONCE (shared by both N-tiles).
        // 16-bit A layout (ISA 7.12.2): per lane-half g, element j of the
        // fragment covers two contiguous 8-float runs: [k0+8g, +8) and
        // [k0+16+8g, +8)  -> 4x float4 (b128) loads per K-step.
        v16h afr[4];
#pragma unroll
        for (int ks = 0; ks < 4; ++ks) {
            const int k0 = ks * 32;
            const float4 r0 = *(const float4*)(xr + k0 + 8 * g);
            const float4 r1 = *(const float4*)(xr + k0 + 8 * g + 4);
            const float4 r2 = *(const float4*)(xr + k0 + 16 + 8 * g);
            const float4 r3 = *(const float4*)(xr + k0 + 16 + 8 * g + 4);
            v16h a;
            a[0]  = (_Float16)r0.x; a[1]  = (_Float16)r0.y;
            a[2]  = (_Float16)r0.z; a[3]  = (_Float16)r0.w;
            a[4]  = (_Float16)r1.x; a[5]  = (_Float16)r1.y;
            a[6]  = (_Float16)r1.z; a[7]  = (_Float16)r1.w;
            a[8]  = (_Float16)r2.x; a[9]  = (_Float16)r2.y;
            a[10] = (_Float16)r2.z; a[11] = (_Float16)r2.w;
            a[12] = (_Float16)r3.x; a[13] = (_Float16)r3.y;
            a[14] = (_Float16)r3.z; a[15] = (_Float16)r3.w;
            afr[ks] = a;
        }

#pragma unroll
        for (int nt = 0; nt < 2; ++nt) {
            v8f acc = {};
#pragma unroll
            for (int ks = 0; ks < 4; ++ks) {
                const int k0 = ks * 32;
                v16h bf;
#pragma unroll
                for (int j = 0; j < 16; ++j) {     // B: lane-half g covers K 8g..8g+7 and 16+8g..16+8g+7
                    const int k = k0 + (j < 8 ? 8 * g + j : 16 + 8 * g + (j - 8));
                    bf[j] = W1h[k * 32 + nt * 16 + ln];
                }
                acc = __builtin_amdgcn_wmma_f32_16x16x32_f16(
                          false, afr[ks], false, bf, (short)0, acc, false, false);
            }
            // epilogue: bias + relu -> LDS (C layout: VGPR r -> row r+8g, col ln)
            const float bias = b1[nt * 16 + ln];
#pragma unroll
            for (int r = 0; r < 8; ++r) {
                const int m = tile * 16 + r + 8 * g;
                float v = acc[r] + bias;
                v = v > 0.f ? v : 0.f;
                h1h[m * H1_STRIDE + nt * 16 + ln] = (_Float16)v;
            }
        }
    }

    // ---- Layer 2 (WMMA, K=32, N=16) ----
#pragma unroll
    for (int tt = 0; tt < 2; ++tt) {
        const int tile = wave * 2 + tt;
        const int row  = tile * 16 + ln;
        v16h a, bf;
#pragma unroll
        for (int j = 0; j < 16; ++j) {
            const int v = j >> 1, w = j & 1;
            const int k = (v < 4 ? 2 * v : 16 + 2 * (v - 4)) + 8 * g + w;
            a[j] = h1h[row * H1_STRIDE + k];
        }
#pragma unroll
        for (int j = 0; j < 16; ++j) {
            const int k = (j < 8 ? 8 * g + j : 16 + 8 * g + (j - 8));
            bf[j] = W2h[k * 16 + ln];
        }
        v8f acc = {};
        acc = __builtin_amdgcn_wmma_f32_16x16x32_f16(
                  false, a, false, bf, (short)0, acc, false, false);
        const float bias = b2[ln];
#pragma unroll
        for (int r = 0; r < 8; ++r) {
            const int m = tile * 16 + r + 8 * g;
            float v = acc[r] + bias;
            v = v > 0.f ? v : 0.f;
            h2f[m * H2_STRIDE + ln] = v;
        }
    }
    __syncthreads();

    // ---- per-sample scalar tail: layer 3, quantum circuit, layers 4/5 ----
    const int s = blockBase + tid;
    if (s >= B) return;

    float hv[16];
#pragma unroll
    for (int i = 0; i < 16; ++i) hv[i] = h2f[tid * H2_STRIDE + i];

    float ang[NQ];
#pragma unroll
    for (int j = 0; j < NQ; ++j) {
        float a = b3[j];
#pragma unroll
        for (int i = 0; i < 16; ++i) a += hv[i] * W3[i * NQ + j];
        // tanh in [-1,1] => clip is identity; scale by pi
        ang[j] = tanhf(a) * 3.14159265358979323846f;
    }

    // quantum state (registers only; all indices compile-time)
    float sr[DIM], si[DIM];
    {   // initial RY layer on |0...0> == product state
        float cq[NQ], sq[NQ];
#pragma unroll
        for (int j = 0; j < NQ; ++j) {
            const float h = ang[j] * 0.5f;
            cq[j] = __cosf(h); sq[j] = __sinf(h);
        }
#pragma unroll
        for (int k = 0; k < DIM; ++k) {
            float amp = 1.f;
#pragma unroll
            for (int q = 0; q < NQ; ++q)
                amp *= ((k >> (5 - q)) & 1) ? sq[q] : cq[q];
            sr[k] = amp; si[k] = 0.f;
        }
    }

    float cc, ss;
#define ANG(idx)  { const float hh = qw[idx] * 0.5f; cc = __cosf(hh); ss = __sinf(hh); }
#define RXRZ(Q)   ANG(0 * NQ + Q); gate_rx<Q>(sr, si, cc, ss); \
                  ANG(1 * NQ + Q); gate_rz<Q>(sr, si, cc, ss);
    RXRZ(0) RXRZ(1) RXRZ(2) RXRZ(3) RXRZ(4) RXRZ(5)

    gate_cnot<0,1>(sr, si); gate_cnot<1,2>(sr, si); gate_cnot<2,3>(sr, si);
    gate_cnot<3,4>(sr, si); gate_cnot<4,5>(sr, si); gate_cnot<5,0>(sr, si);

#define RYRZ(Q)   ANG(2 * NQ + Q); gate_ry<Q>(sr, si, cc, ss); \
                  ANG(3 * NQ + Q); gate_rz<Q>(sr, si, cc, ss);
    RYRZ(0) RYRZ(1) RYRZ(2) RYRZ(3) RYRZ(4) RYRZ(5)

    gate_cnot<0,1>(sr, si); gate_cnot<2,3>(sr, si); gate_cnot<4,5>(sr, si);

#define RXO(Q)    ANG(4 * NQ + Q); gate_rx<Q>(sr, si, cc, ss);
    RXO(0) RXO(1) RXO(2) RXO(3) RXO(4) RXO(5)

    // PauliZ expectations
    float ez[NQ] = {0.f, 0.f, 0.f, 0.f, 0.f, 0.f};
#pragma unroll
    for (int k = 0; k < DIM; ++k) {
        const float p = sr[k] * sr[k] + si[k] * si[k];
#pragma unroll
        for (int q = 0; q < NQ; ++q)
            ez[q] += ((k >> (5 - q)) & 1) ? -p : p;
    }

    // layer 4: relu(ez @ W4 + b4)
    float h4[16];
#pragma unroll
    for (int j = 0; j < 16; ++j) {
        float a = b4[j];
#pragma unroll
        for (int i = 0; i < NQ; ++i) a += ez[i] * W4[i * 16 + j];
        h4[j] = a > 0.f ? a : 0.f;
    }
    // layer 5: h4 @ W5 + b5 -> out (B, 20)
#pragma unroll
    for (int j = 0; j < 20; ++j) {
        float a = b5[j];
#pragma unroll
        for (int i = 0; i < 16; ++i) a += h4[i] * W5[i * 20 + j];
        out[(size_t)s * 20 + j] = a;
    }
}

extern "C" void kernel_launch(void* const* d_in, const int* in_sizes, int n_in,
                              void* d_out, int out_size, void* d_ws, size_t ws_size,
                              hipStream_t stream) {
    const float* x  = (const float*)d_in[0];
    const float* W1 = (const float*)d_in[1];
    const float* b1 = (const float*)d_in[2];
    const float* W2 = (const float*)d_in[3];
    const float* b2 = (const float*)d_in[4];
    const float* W3 = (const float*)d_in[5];
    const float* b3 = (const float*)d_in[6];
    const float* qw = (const float*)d_in[7];
    const float* W4 = (const float*)d_in[8];
    const float* b4 = (const float*)d_in[9];
    const float* W5 = (const float*)d_in[10];
    const float* b5 = (const float*)d_in[11];

    const int B = in_sizes[0] / 128;
    const int grid = (B + 255) / 256;
    qcnn_fused<<<grid, 256, 0, stream>>>(x, W1, b1, W2, b2, W3, b3, qw,
                                         W4, b4, W5, b5, (float*)d_out, B);
}
